// MoE_22883585753530
// MI455X (gfx1250) — matrix operand order repaired and checked by reading, not compiled, not run on hardware
//
#include <hip/hip_runtime.h>
#include <hip/hip_bf16.h>
#include <math.h>

#define NUM_EXPERTS 8
#define H_DIM 1024
#define I_DIM 2048

typedef __attribute__((ext_vector_type(16))) __bf16 v16bf;
typedef __attribute__((ext_vector_type(8)))  float  v8f;

// ---------- WMMA fragment helpers (contiguous k = 16*half + s order, used for both A and B) ----------
// A-matrix 16x32 bf16: lanes 0-15 -> M=lane, K = 0..15
//                      lanes 16-31 -> M=lane-16, K = 16..31
__device__ __forceinline__ v16bf load_a_frag(const __bf16* rowp, int half) {
    const int kbase = half << 4;  // 0 or 16
    v16bf a;
#pragma unroll
    for (int p = 0; p < 16; ++p) a[p] = rowp[kbase + p];
    return a;
}

// B-matrix 32x16 bf16: lane n=lane&15 holds column n; lanes 0-15 K=0..15,
// lanes 16-31 K=16..31 (caller adds half*16).
// fp32 source: 16 consecutive floats, converted (fallback path)
__device__ __forceinline__ v16bf load_b_frag_f32(const float* p) {
    v16bf b;
#pragma unroll
    for (int j = 0; j < 16; ++j) b[j] = (__bf16)p[j];
    return b;
}
// bf16 source: 16 consecutive bf16 = two 16B vector loads, no conversion
__device__ __forceinline__ v16bf load_b_frag_bf16(const __bf16* p) {
    v16bf b;
    ((uint4*)&b)[0] = ((const uint4*)p)[0];
    ((uint4*)&b)[1] = ((const uint4*)p)[1];
    return b;
}

// ---------- kernel 0: zero output + counters ----------
__global__ void moe_zero_kernel(float* __restrict__ out, long long n, int* __restrict__ counts) {
    long long i = (long long)blockIdx.x * blockDim.x + threadIdx.x;
    if (i < n) out[i] = 0.0f;
    if (i < NUM_EXPERTS) counts[i] = 0;
}

// ---------- kernel 0b: fp32 -> bf16 weight conversion (once per call) ----------
__global__ __launch_bounds__(256)
void moe_cvt_kernel(const float* __restrict__ src, __bf16* __restrict__ dst, long long n4) {
    long long i = (long long)blockIdx.x * blockDim.x + threadIdx.x;
    if (i >= n4) return;
    const float4 v = ((const float4*)src)[i];
    __bf16* d = dst + i * 4;
    d[0] = (__bf16)v.x; d[1] = (__bf16)v.y; d[2] = (__bf16)v.z; d[3] = (__bf16)v.w;
}

// ---------- kernel 1: routing (top-2 + softmax, scatter to expert lists) ----------
__global__ __launch_bounds__(256)
void moe_route_kernel(const float* __restrict__ x, const float* __restrict__ w_gate,
                      int* __restrict__ counts, int* __restrict__ tok_ids,
                      float* __restrict__ gates, int T) {
    __shared__ float wg[NUM_EXPERTS * H_DIM];
    for (int i = threadIdx.x; i < NUM_EXPERTS * H_DIM / 4; i += blockDim.x)
        ((float4*)wg)[i] = ((const float4*)w_gate)[i];
    __syncthreads();

    const int t = blockIdx.x * blockDim.x + threadIdx.x;
    if (t >= T) return;

    const float4* xr = (const float4*)(x + (size_t)t * H_DIM);
    float logit[NUM_EXPERTS];
#pragma unroll
    for (int e = 0; e < NUM_EXPERTS; ++e) logit[e] = 0.0f;

    for (int j = 0; j < H_DIM / 4; ++j) {
        const float4 xv = xr[j];
#pragma unroll
        for (int e = 0; e < NUM_EXPERTS; ++e) {
            const float4 wv = ((const float4*)(wg + e * H_DIM))[j];
            logit[e] += xv.x * wv.x + xv.y * wv.y + xv.z * wv.z + xv.w * wv.w;
        }
    }

    int e1 = 0; float v1 = logit[0];
#pragma unroll
    for (int e = 1; e < NUM_EXPERTS; ++e)
        if (logit[e] > v1) { v1 = logit[e]; e1 = e; }
    int e2 = -1; float v2 = -3.0e38f;
#pragma unroll
    for (int e = 0; e < NUM_EXPERTS; ++e)
        if (e != e1 && logit[e] > v2) { v2 = logit[e]; e2 = e; }

    const float ex = __expf(v2 - v1);       // v1 >= v2, numerically safe
    const float g1 = 1.0f / (1.0f + ex);
    const float g2 = ex / (1.0f + ex);

    int s1 = atomicAdd(&counts[e1], 1);
    tok_ids[(size_t)e1 * T + s1] = t;  gates[(size_t)e1 * T + s1] = g1;
    int s2 = atomicAdd(&counts[e2], 1);
    tok_ids[(size_t)e2 * T + s2] = t;  gates[(size_t)e2 * T + s2] = g2;
}

// ---------- kernel 2: fused expert MLP (fc -> gelu -> proj), WMMA bf16 ----------
// Block: 256 threads (8 wave32 waves). Tile: 32 gathered tokens x full H output.
// I processed in chunks of 128; hmid chunk shared through LDS.
// PRECVT=true: weights pre-converted to bf16 in workspace (no inner-loop cvt).
template <bool PRECVT>
__global__ __launch_bounds__(256)
void moe_expert_kernel(const float* __restrict__ x,
                       const float* __restrict__ wfc_f,  const __bf16* __restrict__ wfc_b,
                       const float* __restrict__ b_fc,
                       const float* __restrict__ wpj_f,  const __bf16* __restrict__ wpj_b,
                       const float* __restrict__ b_proj,
                       const int* __restrict__ counts, const int* __restrict__ tok_ids,
                       const float* __restrict__ gates,
                       float* __restrict__ out, int T) {
    const int e = blockIdx.y;
    const int cnt = counts[e];
    const int tile0 = blockIdx.x * 32;
    if (tile0 >= cnt) return;              // uniform early exit

    constexpr int XP = H_DIM + 16;         // x tile row pitch (bf16 elems)
    constexpr int HP = 128 + 8;            // hmid chunk row pitch
    extern __shared__ char smem[];
    __bf16* x_lds = (__bf16*)smem;                                   // 32*XP
    __bf16* h_lds = (__bf16*)(smem + (size_t)32 * XP * 2);           // 32*HP
    int*    tk    = (int*)  (smem + (size_t)32 * XP * 2 + 32 * HP * 2);
    float*  gt    = (float*)((char*)tk + 32 * sizeof(int));

    const int tid  = threadIdx.x;
    const int lane = tid & 31;
    const int w    = tid >> 5;
    const int nlo  = lane & 15;
    const int half = lane >> 4;

    if (tid < 32) {
        const int s = tile0 + tid;
        const bool valid = s < cnt;
        tk[tid] = tok_ids[(size_t)e * T + (valid ? s : 0)];
        gt[tid] = valid ? gates[(size_t)e * T + s] : 0.0f;   // gate 0 => padded row
    }
    __syncthreads();

    // Stage gathered x rows into LDS as bf16
    for (int idx = tid; idx < 32 * (H_DIM / 4); idx += 256) {
        const int row = idx >> 8;          // H_DIM/4 == 256
        const int c4  = idx & 255;
        const float4 xv = ((const float4*)(x + (size_t)tk[row] * H_DIM))[c4];
        __bf16* d = x_lds + (size_t)row * XP + c4 * 4;
        d[0] = (__bf16)xv.x; d[1] = (__bf16)xv.y; d[2] = (__bf16)xv.z; d[3] = (__bf16)xv.w;
    }
    __syncthreads();

    v8f acc[16];
#pragma unroll
    for (int i = 0; i < 16; ++i) acc[i] = (v8f){0.f,0.f,0.f,0.f,0.f,0.f,0.f,0.f};

    const int mtH     = w >> 2;            // hmid M-tile owned by this wave
    const int ntiBase = (w & 3) * 2;       // 2 hmid N(i)-tiles owned by this wave

    const float*  wfcF = wfc_f + (size_t)e * I_DIM * H_DIM;
    const __bf16* wfcB = wfc_b + (size_t)e * I_DIM * H_DIM;
    const float*  wpjF = wpj_f + (size_t)e * H_DIM * I_DIM;
    const __bf16* wpjB = wpj_b + (size_t)e * H_DIM * I_DIM;

    for (int c = 0; c < I_DIM / 128; ++c) {
        const int cbase = c * 128;

        // ---- Phase A: hmid chunk = gelu(x @ w_fc^T + b_fc)  (2 tiles / wave) ----
#pragma unroll
        for (int t2 = 0; t2 < 2; ++t2) {
            const int nti = ntiBase + t2;
            const int ig  = cbase + nti * 16 + nlo;            // this lane's i column
            const __bf16* arow = x_lds + (size_t)(mtH * 16 + nlo) * XP;
            const float*  browF = wfcF + (size_t)ig * H_DIM + half * 16;
            const __bf16* browB = wfcB + (size_t)ig * H_DIM + half * 16;
            v8f hacc = (v8f){0.f,0.f,0.f,0.f,0.f,0.f,0.f,0.f};
#pragma unroll 4
            for (int kk = 0; kk < H_DIM / 32; ++kk) {
                v16bf a = load_a_frag(arow + kk * 32, half);
                v16bf b;
                if constexpr (PRECVT) {
                    __builtin_prefetch(browB + kk * 32 + 256, 0, 0);
                    b = load_b_frag_bf16(browB + kk * 32);
                } else {
                    __builtin_prefetch(browF + kk * 32 + 128, 0, 0);
                    b = load_b_frag_f32(browF + kk * 32);
                }
                hacc = __builtin_amdgcn_wmma_f32_16x16x32_bf16(
                    false, a, false, b, (short)0, hacc, false, false);
            }
            const float bias = b_fc[(size_t)e * I_DIM + cbase + nti * 16 + nlo];
            __bf16* hdst = h_lds + nti * 16 + nlo;
#pragma unroll
            for (int r = 0; r < 8; ++r) {
                float v = hacc[r] + bias;
                v = 0.5f * v * (1.0f + erff(v * 0.70710678118f));   // exact gelu
                hdst[(size_t)(mtH * 16 + r + 8 * half) * HP] = (__bf16)v;
            }
        }
        __syncthreads();

        // ---- Phase B: y += hmid_chunk @ w_proj_chunk^T  (16 tiles / wave) ----
#pragma unroll
        for (int mt = 0; mt < 2; ++mt) {
            const __bf16* arow = h_lds + (size_t)(mt * 16 + nlo) * HP;
#pragma unroll
            for (int nt = 0; nt < 8; ++nt) {
                const int hbase = (w * 8 + nt) * 16;
                v8f a2 = acc[mt * 8 + nt];
#pragma unroll
                for (int k4 = 0; k4 < 4; ++k4) {
                    v16bf a = load_a_frag(arow + k4 * 32, half);
                    v16bf b;
                    if constexpr (PRECVT) {
                        b = load_b_frag_bf16(wpjB + (size_t)(hbase + nlo) * I_DIM
                                             + cbase + half * 16 + k4 * 32);
                    } else {
                        b = load_b_frag_f32(wpjF + (size_t)(hbase + nlo) * I_DIM
                                            + cbase + half * 16 + k4 * 32);
                    }
                    a2 = __builtin_amdgcn_wmma_f32_16x16x32_bf16(
                        false, a, false, b, (short)0, a2, false, false);
                }
                acc[mt * 8 + nt] = a2;
            }
        }
        __syncthreads();
    }

    // ---- Epilogue: out[token][h] += gate * (y + b_proj) ----
#pragma unroll
    for (int mt = 0; mt < 2; ++mt) {
#pragma unroll
        for (int nt = 0; nt < 8; ++nt) {
            const int h = (w * 8 + nt) * 16 + nlo;
            const float bias = b_proj[(size_t)e * H_DIM + h];
            v8f a2 = acc[mt * 8 + nt];
#pragma unroll
            for (int r = 0; r < 8; ++r) {
                const int m = mt * 16 + r + 8 * half;
                const float g = gt[m];
                if (g != 0.0f)
                    atomicAdd(out + (size_t)tk[m] * H_DIM + h, (a2[r] + bias) * g);
            }
        }
    }
}

extern "C" void kernel_launch(void* const* d_in, const int* in_sizes, int n_in,
                              void* d_out, int out_size, void* d_ws, size_t ws_size,
                              hipStream_t stream) {
    const float* x      = (const float*)d_in[0];
    const float* w_gate = (const float*)d_in[1];
    const float* w_fc   = (const float*)d_in[2];
    const float* b_fc   = (const float*)d_in[3];
    const float* w_proj = (const float*)d_in[4];
    const float* b_proj = (const float*)d_in[5];
    float* out = (float*)d_out;

    const int T = in_sizes[0] / H_DIM;   // 16384 tokens
    const long long WN = (long long)NUM_EXPERTS * I_DIM * H_DIM;   // elems per weight tensor

    // workspace layout
    size_t off = 0;
    int*   counts  = (int*)d_ws;                          off = 128;
    int*   tok_ids = (int*)((char*)d_ws + off);           off += (size_t)NUM_EXPERTS * T * sizeof(int);
    float* gates   = (float*)((char*)d_ws + off);         off += (size_t)NUM_EXPERTS * T * sizeof(float);
    off = (off + 255) & ~(size_t)255;
    __bf16* wfc_b  = (__bf16*)((char*)d_ws + off);        off += (size_t)WN * sizeof(__bf16);
    __bf16* wpj_b  = (__bf16*)((char*)d_ws + off);        off += (size_t)WN * sizeof(__bf16);
    const bool precvt = ws_size >= off;   // deterministic: fixed per harness config

    const long long n = (long long)out_size;
    moe_zero_kernel<<<(unsigned)((n + 255) / 256), 256, 0, stream>>>(out, n, counts);
    moe_route_kernel<<<(T + 255) / 256, 256, 0, stream>>>(x, w_gate, counts, tok_ids, gates, T);

    dim3 grid(T / 32, NUM_EXPERTS);
    const size_t lds = (size_t)(32 * (H_DIM + 16) + 32 * (128 + 8)) * sizeof(__bf16)
                     + 32 * sizeof(int) + 32 * sizeof(float) + 256;

    if (precvt) {
        const long long n4 = WN / 4;
        moe_cvt_kernel<<<(unsigned)((n4 + 255) / 256), 256, 0, stream>>>(w_fc, wfc_b, n4);
        moe_cvt_kernel<<<(unsigned)((n4 + 255) / 256), 256, 0, stream>>>(w_proj, wpj_b, n4);
        moe_expert_kernel<true><<<grid, 256, lds, stream>>>(
            x, w_fc, wfc_b, b_fc, w_proj, wpj_b, b_proj,
            counts, tok_ids, gates, out, T);
    } else {
        moe_expert_kernel<false><<<grid, 256, lds, stream>>>(
            x, w_fc, wfc_b, b_fc, w_proj, wpj_b, b_proj,
            counts, tok_ids, gates, out, T);
    }
}
